// TransformerBlock_62646392979614
// MI455X (gfx1250) — matrix operand order repaired and checked by reading, not compiled, not run on hardware
//
#include <hip/hip_runtime.h>
#include <hip/hip_bf16.h>

// ---------------------------------------------------------------------------
// Types for CDNA5 WMMA (gfx1250, wave32)
// ---------------------------------------------------------------------------
typedef __attribute__((ext_vector_type(16))) __bf16          v16bf;
typedef __attribute__((ext_vector_type(8)))  float           v8f;
typedef __attribute__((ext_vector_type(8)))  unsigned short  v8u;
typedef __attribute__((ext_vector_type(4)))  int             v4i;

// gfx1250 async global->LDS copy path (ASYNCcnt tracked), with safe fallback.
#if defined(__HIP_DEVICE_COMPILE__) && \
    __has_builtin(__builtin_amdgcn_global_load_async_to_lds_b128) && \
    __has_builtin(__builtin_amdgcn_s_wait_asynccnt)
#define ASYNC_LDS 1
#else
#define ASYNC_LDS 0
#endif

#if ASYNC_LDS
#define WAIT_ASYNC(n) __builtin_amdgcn_s_wait_asynccnt(n)
typedef __attribute__((address_space(1))) v4i* as1_v4i_p;
typedef __attribute__((address_space(3))) v4i* as3_v4i_p;
#else
#define WAIT_ASYNC(n) ((void)0)
#endif

// Copy 16 bytes global -> LDS (async when available)
__device__ __forceinline__ void copy16(const unsigned short* g, unsigned short* l) {
#if ASYNC_LDS
    __builtin_amdgcn_global_load_async_to_lds_b128(
        (as1_v4i_p)(g), (as3_v4i_p)(l), 0, 0);
#else
    *(v8u*)l = *(const v8u*)g;
#endif
}

// fp32 -> bf16 (round to nearest even)
__device__ __forceinline__ unsigned short f2bf(float f) {
    unsigned int u = __float_as_uint(f);
    u += 0x7fffu + ((u >> 16) & 1u);
    return (unsigned short)(u >> 16);
}

// A-matrix fragment (16x32 bf16), per ISA 7.12.2:
//   lanes 0-15 : VGPR0-3 = K0..7,  VGPR4-7 = K16..23   (row M = lane&15)
//   lanes16-31 : VGPR0-3 = K8..15, VGPR4-7 = K24..31
__device__ __forceinline__ v16bf frag_a(const unsigned short* p, int lane) {
    int klo = (lane & 16) ? 8 : 0;
    union { v16bf bf; v8u u[2]; } r;
    r.u[0] = *(const v8u*)(p + klo);
    r.u[1] = *(const v8u*)(p + klo + 16);
    return r.bf;
}

// B-matrix fragment (32x16 bf16), per ISA 7.12.5 pattern:
//   lanes 0-15 : K0..7, K16..23;  lanes16-31 : K8..15, K24..31  (col N = lane&15)
__device__ __forceinline__ v16bf frag_b(const unsigned short* p, int lane) {
    int klo = (lane & 16) ? 8 : 0;
    union { v16bf bf; v8u u[2]; } r;
    r.u[0] = *(const v8u*)(p + klo);
    r.u[1] = *(const v8u*)(p + klo + 16);
    return r.bf;
}

__device__ __forceinline__ v8f wmma_bf16(v16bf a, v16bf b, v8f c) {
    return __builtin_amdgcn_wmma_f32_16x16x32_bf16(false, a, false, b,
                                                   (short)0, c, false, false);
}

// ---------------------------------------------------------------------------
// Weight convert + transpose:  W (K x N fp32, row major)  ->  Wt (N x K bf16)
// block (32,8), grid (N/32, K/32)
// ---------------------------------------------------------------------------
__global__ void wtrans_kernel(const float* __restrict__ W,
                              unsigned short* __restrict__ Wt, int K, int N) {
    __shared__ float tile[32][33];
    int n0 = blockIdx.x * 32, k0 = blockIdx.y * 32;
    int tx = threadIdx.x, ty = threadIdx.y;
#pragma unroll
    for (int i = 0; i < 4; ++i)
        tile[ty + i * 8][tx] = W[(size_t)(k0 + ty + i * 8) * N + n0 + tx];
    __syncthreads();
#pragma unroll
    for (int i = 0; i < 4; ++i)
        Wt[(size_t)(n0 + ty + i * 8) * K + k0 + tx] = f2bf(tile[tx][ty + i * 8]);
}

// ---------------------------------------------------------------------------
// LayerNorm over D=1024, fp32 in -> bf16 out.  block 256, grid = rows
// ---------------------------------------------------------------------------
__global__ void ln_bf16_kernel(const float* __restrict__ x,
                               const float* __restrict__ g,
                               const float* __restrict__ b,
                               unsigned short* __restrict__ out) {
    int row = blockIdx.x;
    int tid = threadIdx.x;
    const float* xr = x + (size_t)row * 1024;
    float v[4];
    float s = 0.f, ss = 0.f;
#pragma unroll
    for (int kk = 0; kk < 4; ++kk) {
        v[kk] = xr[tid + kk * 256];
        s += v[kk]; ss += v[kk] * v[kk];
    }
#pragma unroll
    for (int m = 16; m >= 1; m >>= 1) {
        s  += __shfl_xor(s, m);
        ss += __shfl_xor(ss, m);
    }
    __shared__ float rs[8], rss[8];
    int wid = tid >> 5, lane = tid & 31;
    if (lane == 0) { rs[wid] = s; rss[wid] = ss; }
    __syncthreads();
    if (wid == 0) {
        float a  = (lane < 8) ? rs[lane]  : 0.f;
        float a2 = (lane < 8) ? rss[lane] : 0.f;
#pragma unroll
        for (int m = 4; m >= 1; m >>= 1) {
            a  += __shfl_xor(a, m);
            a2 += __shfl_xor(a2, m);
        }
        if (lane == 0) { rs[0] = a; rss[0] = a2; }
    }
    __syncthreads();
    float mean = rs[0] * (1.f / 1024.f);
    float var  = rss[0] * (1.f / 1024.f) - mean * mean;
    float rstd = rsqrtf(var + 1e-5f);
#pragma unroll
    for (int kk = 0; kk < 4; ++kk) {
        int i = tid + kk * 256;
        out[(size_t)row * 1024 + i] = f2bf((v[kk] - mean) * rstd * g[i] + b[i]);
    }
}

// ---------------------------------------------------------------------------
// WMMA GEMM: C[M,N] = A[M,K](bf16 row-major) * Bt[N,K](bf16 N-major) + bias
// optional residual add, exact GELU, bf16 or fp32 output.
// block = 256 (8 waves), block tile 128x128, wave tile 64x32.
// K-step 64, double-buffered LDS, async global->LDS copies.
// ---------------------------------------------------------------------------
template <bool GELU, bool RES, bool OBF>
__global__ void gemm_wmma(const unsigned short* __restrict__ A,
                          const unsigned short* __restrict__ Bt,
                          const float* __restrict__ bias,
                          const float* __restrict__ res,
                          float* __restrict__ outf,
                          unsigned short* __restrict__ outb,
                          int M, int N, int K) {
    __shared__ __align__(16) unsigned short As[2][128 * 72];
    __shared__ __align__(16) unsigned short Bs[2][128 * 72];

    int tid = threadIdx.x, wid = tid >> 5, lane = tid & 31;
    int m0 = blockIdx.y * 128, n0 = blockIdx.x * 128;
    int wm = (wid >> 2) * 64, wn = (wid & 3) * 32;

    v8f zero = {0.f, 0.f, 0.f, 0.f, 0.f, 0.f, 0.f, 0.f};
    v8f acc[4][2];
#pragma unroll
    for (int mi = 0; mi < 4; ++mi)
#pragma unroll
        for (int ni = 0; ni < 2; ++ni) acc[mi][ni] = zero;

    // 128 rows x 64 halves per matrix -> 1024 16B chunks, 4 per thread
    auto load_tile = [&](int buf, int kk) {
#pragma unroll
        for (int i = tid; i < 1024; i += 256) {
            int r = i >> 3, c = (i & 7) << 3;
            copy16(&A[(size_t)(m0 + r) * K + kk + c], &As[buf][r * 72 + c]);
            copy16(&Bt[(size_t)(n0 + r) * K + kk + c], &Bs[buf][r * 72 + c]);
        }
    };

    int nk = K >> 6;
    load_tile(0, 0);
    for (int kt = 0; kt < nk; ++kt) {
        int buf = kt & 1;
        if (kt + 1 < nk) {
            load_tile(buf ^ 1, (kt + 1) << 6);
            WAIT_ASYNC(8);      // this wave's tile-kt copies (older) complete
        } else {
            WAIT_ASYNC(0);
        }
        __syncthreads();        // all waves' tile-kt data visible in LDS

#pragma unroll
        for (int ks = 0; ks < 2; ++ks) {
            v16bf afr[4], bfr[2];
#pragma unroll
            for (int mi = 0; mi < 4; ++mi)
                afr[mi] = frag_a(&As[buf][(wm + mi * 16 + (lane & 15)) * 72 + ks * 32], lane);
#pragma unroll
            for (int ni = 0; ni < 2; ++ni)
                bfr[ni] = frag_b(&Bs[buf][(wn + ni * 16 + (lane & 15)) * 72 + ks * 32], lane);
#pragma unroll
            for (int mi = 0; mi < 4; ++mi)
#pragma unroll
                for (int ni = 0; ni < 2; ++ni)
                    acc[mi][ni] = wmma_bf16(afr[mi], bfr[ni], acc[mi][ni]);
        }
        __syncthreads();        // reads of buf done before it is refilled
    }

    // Epilogue.  C layout: VGPR j -> row (j | j+8 for hi half), col = lane&15
#pragma unroll
    for (int mi = 0; mi < 4; ++mi) {
#pragma unroll
        for (int ni = 0; ni < 2; ++ni) {
            int n = n0 + wn + ni * 16 + (lane & 15);
            float bv = bias[n];
#pragma unroll
            for (int j = 0; j < 8; ++j) {
                int m = m0 + wm + mi * 16 + j + ((lane & 16) ? 8 : 0);
                float v = acc[mi][ni][j] + bv;
                if (RES)  v += res[(size_t)m * N + n];
                if (GELU) v = 0.5f * v * (1.0f + erff(v * 0.70710678118654752f));
                if (OBF)  outb[(size_t)m * N + n] = f2bf(v);
                else      outf[(size_t)m * N + n] = v;
            }
        }
    }
}

// ---------------------------------------------------------------------------
// Flash attention, T=2048, Dh=64, scale=1/8.
// grid (T/64, H, B), block 128 (4 waves; wave w owns query rows w*16..w*16+15)
// q/k/v : bf16 (B,T,H*Dh) row-major.  o : bf16 same layout.
// ---------------------------------------------------------------------------
__global__ void attn_kernel(const unsigned short* __restrict__ q,
                            const unsigned short* __restrict__ k,
                            const unsigned short* __restrict__ v,
                            unsigned short* __restrict__ o) {
    const int T = 2048, D = 1024, Dh = 64;
    __shared__ __align__(16) unsigned short Qs[64 * 72];
    __shared__ __align__(16) unsigned short Ks[64 * 72];
    __shared__ __align__(16) unsigned short Vt[64 * 72];
    __shared__ __align__(16) unsigned short Ps[4 * 16 * 72];

    int tid = threadIdx.x, wid = tid >> 5, lane = tid & 31;
    int qt0 = blockIdx.x * 64;
    int h   = blockIdx.y;
    int bidx = blockIdx.z;
    size_t base = ((size_t)bidx * T) * D + h * Dh;

    // Load Q tile (64 x 64) once (async)
    for (int i = tid; i < 512; i += 128) {
        int r = i >> 3, c = (i & 7) << 3;
        copy16(&q[base + (size_t)(qt0 + r) * D + c], &Qs[r * 72 + c]);
    }

    v8f zero = {0.f, 0.f, 0.f, 0.f, 0.f, 0.f, 0.f, 0.f};
    v8f oacc[4];
#pragma unroll
    for (int ni = 0; ni < 4; ++ni) oacc[ni] = zero;
    float mrun[8], lrun[8];
#pragma unroll
    for (int j = 0; j < 8; ++j) { mrun[j] = -3.0e38f; lrun[j] = 0.f; }

    const float SCALE = 0.125f; // 1/sqrt(64)

    for (int kb = 0; kb < T / 64; ++kb) {
        int kt0 = kb * 64;
        __syncthreads(); // previous iteration's reads done
        for (int i = tid; i < 512; i += 128) {
            int r = i >> 3, c = (i & 7) << 3;
            copy16(&k[base + (size_t)(kt0 + r) * D + c], &Ks[r * 72 + c]);
        }
        // V transposed (Vt[dh][key]) for B-operand of P@V
        for (int i = tid; i < 4096; i += 128) {
            int r = i >> 6, c = i & 63;
            Vt[c * 72 + r] = v[base + (size_t)(kt0 + r) * D + c];
        }
        WAIT_ASYNC(0);
        __syncthreads();

        // S = Q @ K^T   (wave tile 16 x 64)
        v8f sacc[4];
#pragma unroll
        for (int ni = 0; ni < 4; ++ni) sacc[ni] = zero;
        const unsigned short* qrow = &Qs[(wid * 16 + (lane & 15)) * 72];
#pragma unroll
        for (int ks = 0; ks < 2; ++ks) {
            v16bf a = frag_a(qrow + ks * 32, lane);
#pragma unroll
            for (int ni = 0; ni < 4; ++ni) {
                v16bf bf = frag_b(&Ks[(ni * 16 + (lane & 15)) * 72 + ks * 32], lane);
                sacc[ni] = wmma_bf16(a, bf, sacc[ni]);
            }
        }

        // Online softmax update (per VGPR j: row j in lanes 0-15, row j+8 in 16-31)
#pragma unroll
        for (int j = 0; j < 8; ++j) {
            float s0 = sacc[0][j] * SCALE, s1 = sacc[1][j] * SCALE;
            float s2 = sacc[2][j] * SCALE, s3 = sacc[3][j] * SCALE;
            float mb = fmaxf(fmaxf(s0, s1), fmaxf(s2, s3));
#pragma unroll
            for (int m = 8; m >= 1; m >>= 1) mb = fmaxf(mb, __shfl_xor(mb, m));
            float mnew = fmaxf(mrun[j], mb);
            float p0 = __expf(s0 - mnew), p1 = __expf(s1 - mnew);
            float p2 = __expf(s2 - mnew), p3 = __expf(s3 - mnew);
            float ls = p0 + p1 + p2 + p3;
#pragma unroll
            for (int m = 8; m >= 1; m >>= 1) ls += __shfl_xor(ls, m);
            float alpha = __expf(mrun[j] - mnew);
            lrun[j] = lrun[j] * alpha + ls;
            mrun[j] = mnew;
#pragma unroll
            for (int ni = 0; ni < 4; ++ni) oacc[ni][j] *= alpha;
            int prow = j + ((lane & 16) ? 8 : 0);
            unsigned short* pp = &Ps[(wid * 16 + prow) * 72 + (lane & 15)];
            pp[0] = f2bf(p0); pp[16] = f2bf(p1); pp[32] = f2bf(p2); pp[48] = f2bf(p3);
        }

        // O += P @ V  (same-wave LDS, in-order, no barrier needed)
        const unsigned short* prow2 = &Ps[(wid * 16 + (lane & 15)) * 72];
#pragma unroll
        for (int ks = 0; ks < 2; ++ks) {
            v16bf a = frag_a(prow2 + ks * 32, lane);
#pragma unroll
            for (int ni = 0; ni < 4; ++ni) {
                v16bf bf = frag_b(&Vt[(ni * 16 + (lane & 15)) * 72 + ks * 32], lane);
                oacc[ni] = wmma_bf16(a, bf, oacc[ni]);
            }
        }
    }

    // Normalize and write bf16 output
#pragma unroll
    for (int ni = 0; ni < 4; ++ni) {
#pragma unroll
        for (int j = 0; j < 8; ++j) {
            int r = wid * 16 + j + ((lane & 16) ? 8 : 0);
            float val = oacc[ni][j] / lrun[j];
            o[base + (size_t)(qt0 + r) * D + ni * 16 + (lane & 15)] = f2bf(val);
        }
    }
}

// ---------------------------------------------------------------------------
// Launcher
// ---------------------------------------------------------------------------
extern "C" void kernel_launch(void* const* d_in, const int* in_sizes, int n_in,
                              void* d_out, int out_size, void* d_ws, size_t ws_size,
                              hipStream_t stream) {
    (void)in_sizes; (void)n_in; (void)out_size; (void)ws_size;
    const int B = 2, T = 2048, D = 1024, F = 4096;
    const int M = B * T; // 4096 rows

    const float* x     = (const float*)d_in[0];
    const float* ln1_g = (const float*)d_in[1];
    const float* ln1_b = (const float*)d_in[2];
    const float* Wq    = (const float*)d_in[3];
    const float* bq    = (const float*)d_in[4];
    const float* Wk    = (const float*)d_in[5];
    const float* bk    = (const float*)d_in[6];
    const float* Wv    = (const float*)d_in[7];
    const float* bv    = (const float*)d_in[8];
    const float* Wo    = (const float*)d_in[9];
    const float* bo    = (const float*)d_in[10];
    const float* ln2_g = (const float*)d_in[11];
    const float* ln2_b = (const float*)d_in[12];
    const float* W1    = (const float*)d_in[13];
    const float* b1    = (const float*)d_in[14];
    const float* W2    = (const float*)d_in[15];
    const float* b2    = (const float*)d_in[16];
    float* out = (float*)d_out;

    // Workspace layout (bytes)
    char* ws = (char*)d_ws;
    const size_t SZ_ACT = (size_t)M * D * 2;      // 8 MB bf16 activation
    unsigned short* xn    = (unsigned short*)(ws + 0 * SZ_ACT);
    unsigned short* qbuf  = (unsigned short*)(ws + 1 * SZ_ACT);
    unsigned short* kbuf  = (unsigned short*)(ws + 2 * SZ_ACT);
    unsigned short* vbuf  = (unsigned short*)(ws + 3 * SZ_ACT);
    unsigned short* abuf  = (unsigned short*)(ws + 4 * SZ_ACT);
    unsigned short* hbuf  = (unsigned short*)(ws + 5 * SZ_ACT);
    size_t off = 6 * SZ_ACT;
    unsigned short* f1buf = (unsigned short*)(ws + off); off += (size_t)M * F * 2; // 32 MB
    float*          x2    = (float*)(ws + off);          off += (size_t)M * D * 4; // 16 MB
    unsigned short* WqT   = (unsigned short*)(ws + off); off += (size_t)D * D * 2;
    unsigned short* WkT   = (unsigned short*)(ws + off); off += (size_t)D * D * 2;
    unsigned short* WvT   = (unsigned short*)(ws + off); off += (size_t)D * D * 2;
    unsigned short* WoT   = (unsigned short*)(ws + off); off += (size_t)D * D * 2;
    unsigned short* W1T   = (unsigned short*)(ws + off); off += (size_t)D * F * 2;
    unsigned short* W2T   = (unsigned short*)(ws + off); off += (size_t)F * D * 2;

    dim3 tb(32, 8);
    // Weight convert + transpose
    wtrans_kernel<<<dim3(D / 32, D / 32), tb, 0, stream>>>(Wq, WqT, D, D);
    wtrans_kernel<<<dim3(D / 32, D / 32), tb, 0, stream>>>(Wk, WkT, D, D);
    wtrans_kernel<<<dim3(D / 32, D / 32), tb, 0, stream>>>(Wv, WvT, D, D);
    wtrans_kernel<<<dim3(D / 32, D / 32), tb, 0, stream>>>(Wo, WoT, D, D);
    wtrans_kernel<<<dim3(F / 32, D / 32), tb, 0, stream>>>(W1, W1T, D, F);
    wtrans_kernel<<<dim3(D / 32, F / 32), tb, 0, stream>>>(W2, W2T, F, D);

    // LN1
    ln_bf16_kernel<<<M, 256, 0, stream>>>(x, ln1_g, ln1_b, xn);

    // QKV projections (bf16 out)
    dim3 gD(D / 128, M / 128);
    gemm_wmma<false, false, true><<<gD, 256, 0, stream>>>(xn, WqT, bq, nullptr, nullptr, qbuf, M, D, D);
    gemm_wmma<false, false, true><<<gD, 256, 0, stream>>>(xn, WkT, bk, nullptr, nullptr, kbuf, M, D, D);
    gemm_wmma<false, false, true><<<gD, 256, 0, stream>>>(xn, WvT, bv, nullptr, nullptr, vbuf, M, D, D);

    // Flash attention
    attn_kernel<<<dim3(T / 64, 16, B), 128, 0, stream>>>(qbuf, kbuf, vbuf, abuf);

    // Output projection + residual (fp32 out to x2)
    gemm_wmma<false, true, false><<<gD, 256, 0, stream>>>(abuf, WoT, bo, x, x2, nullptr, M, D, D);

    // LN2
    ln_bf16_kernel<<<M, 256, 0, stream>>>(x2, ln2_g, ln2_b, hbuf);

    // FFN1 with exact GELU (bf16 out)
    gemm_wmma<true, false, true><<<dim3(F / 128, M / 128), 256, 0, stream>>>(
        hbuf, W1T, b1, nullptr, nullptr, f1buf, M, F, D);

    // FFN2 + residual -> final fp32 output
    gemm_wmma<false, true, false><<<gD, 256, 0, stream>>>(
        f1buf, W2T, b2, x2, out, nullptr, M, D, F);
}
